// SelfAttention2D_46016279609667
// MI455X (gfx1250) — compile-verified, matrix operand order checked by reading, not run on hardware
//
#include <hip/hip_runtime.h>
#include <hip/hip_bf16.h>

// ---------------------------------------------------------------------------
// CDNA5 / gfx1250 bf16-WMMA implementation of the SelfAttention2D block.
// wave32, WGP-mode. All big GEMMs -> v_wmma_f32_16x16x32_bf16, with
// GLOBAL_LOAD_ASYNC_TO_LDS_B128 (ASYNCcnt) staging for the LDS pipeline.
// ---------------------------------------------------------------------------

typedef __bf16 bf16_t;
typedef __attribute__((ext_vector_type(16))) __bf16 v16bf;
typedef __attribute__((ext_vector_type(8)))  __bf16 v8bf;
typedef __attribute__((ext_vector_type(8)))  float  v8f;

union Frag { v16bf v; v8bf h[2]; };

__device__ __forceinline__ v8f wmma_bf16(const Frag& a, const Frag& b, v8f c) {
  return __builtin_amdgcn_wmma_f32_16x16x32_bf16(false, a.v, false, b.v,
                                                 (short)0, c, false, false);
}

__device__ __forceinline__ float waveRedSum(float v) {
#pragma unroll
  for (int o = 16; o > 0; o >>= 1) v += __shfl_xor(v, o, 32);
  return v;
}

// CDNA5 async copy: 16B global -> 16B LDS per lane; INST_OFFSET applies to both
// addresses, so one (lds_off, gaddr) pair covers a 32B chunk with offsets 0/16.
__device__ __forceinline__ void async_copy_32B(unsigned lds_off, const void* gaddr) {
  asm volatile("global_load_async_to_lds_b128 %0, %1, off"
               :: "v"(lds_off), "v"(gaddr) : "memory");
  asm volatile("global_load_async_to_lds_b128 %0, %1, off offset:16"
               :: "v"(lds_off), "v"(gaddr) : "memory");
}

__device__ __forceinline__ void wait_async0() {
  asm volatile("s_wait_asynccnt 0x0" ::: "memory");
}

// Problem constants
#define BB   512
#define CC   2048
#define DD   2048
#define NHH  8
#define DHH  256
#define LL   64
#define HIDD 8192
#define MTOK (BB * LL)   // 32768

// ---------------------------------------------------------------------------
// Elementwise f32 -> bf16 cast (weights)
// ---------------------------------------------------------------------------
__global__ __launch_bounds__(256)
void cast_f32_bf16(const float* __restrict__ in, bf16_t* __restrict__ out, size_t n) {
  size_t i = (size_t)blockIdx.x * 256 + threadIdx.x;
  size_t stride = (size_t)gridDim.x * 256;
  for (; i < n; i += stride) out[i] = (bf16_t)in[i];
}

// ---------------------------------------------------------------------------
// x (B,C,8,8) f32 -> xt (B*L, C) bf16   (LDS tile transpose, coalesced both ways)
// grid (C/64, B), block 256
// ---------------------------------------------------------------------------
__global__ __launch_bounds__(256)
void transpose_x(const float* __restrict__ x, bf16_t* __restrict__ xt) {
  __shared__ float tile[64][65];
  const int b  = blockIdx.y;
  const int c0 = blockIdx.x * 64;
#pragma unroll
  for (int j = 0; j < 16; ++j) {
    int idx = threadIdx.x + j * 256;
    int c = idx >> 6, l = idx & 63;           // read coalesced along l
    tile[c][l] = x[((size_t)b * CC + c0 + c) * LL + l];
  }
  __syncthreads();
#pragma unroll
  for (int j = 0; j < 16; ++j) {
    int idx = threadIdx.x + j * 256;
    int l = idx >> 6, c = idx & 63;           // write coalesced along c
    xt[((size_t)b * LL + l) * CC + c0 + c] = (bf16_t)tile[c][l];
  }
}

// ---------------------------------------------------------------------------
// Tiled bf16 WMMA GEMM:  out[M,N] = A[M,K] @ Bw[N,K]^T (+bias, epilogue)
// Block 256 threads = 8 waves; tile 128x128x32; double-buffered LDS fed by
// GLOBAL_LOAD_ASYNC_TO_LDS_B128 (no VGPR staging).
// ---------------------------------------------------------------------------
#define BM 128
#define BN 128
#define BK 32
#define LDSK 40  // padded row stride in bf16 elements (80B, 16B-aligned)

#define EPI_F32   0  // out f32 = acc + bias
#define EPI_BF16  1  // out bf16 = acc + bias
#define EPI_RESID 2  // out f32 = resid + gamma[0]*(acc + bias)
#define EPI_GELU  3  // out bf16 = gelu_exact(acc + bias)

template <int EPI>
__global__ __launch_bounds__(256)
void gemm_bf16_wmma(const bf16_t* __restrict__ A, const bf16_t* __restrict__ Bw,
                    const float* __restrict__ bias, const float* __restrict__ resid,
                    const float* __restrict__ gamma,
                    float* __restrict__ outF, bf16_t* __restrict__ outB,
                    int M, int N, int K) {
  __shared__ bf16_t sA[2][BM * LDSK];
  __shared__ bf16_t sB[2][BN * LDSK];

  const int tid  = threadIdx.x;
  const int lane = tid & 31;
  const int wave = tid >> 5;
  const int wm   = wave & 3;   // 4 row-groups of 32
  const int wn   = wave >> 2;  // 2 col-groups of 64
  const int blockM = blockIdx.y * BM;
  const int blockN = blockIdx.x * BN;

  const int ld_row  = tid >> 1;        // 0..127
  const int ld_half = (tid & 1) * 16;  // 0 / 16 bf16 within 32-wide K slice

  // LDS byte offsets (wave-relative) for this thread's async destinations
  const unsigned ldsA0 = (unsigned)(uintptr_t)&sA[0][ld_row * LDSK + ld_half];
  const unsigned ldsA1 = (unsigned)(uintptr_t)&sA[1][ld_row * LDSK + ld_half];
  const unsigned ldsB0 = (unsigned)(uintptr_t)&sB[0][ld_row * LDSK + ld_half];
  const unsigned ldsB1 = (unsigned)(uintptr_t)&sB[1][ld_row * LDSK + ld_half];

  const bf16_t* gA = A  + (size_t)(blockM + ld_row) * K + ld_half;
  const bf16_t* gB = Bw + (size_t)(blockN + ld_row) * K + ld_half;

  v8f zero = {};
  v8f acc[2][4];
#pragma unroll
  for (int i = 0; i < 2; ++i)
#pragma unroll
    for (int j = 0; j < 4; ++j) acc[i][j] = zero;

  const int KS = K / BK;

  // prologue: async-fill buffer 0
  async_copy_32B(ldsA0, gA);
  async_copy_32B(ldsB0, gB);
  wait_async0();
  __syncthreads();

  const int fr = lane & 15;
  const int kb = (lane >> 4) * 8;

  for (int ks = 0; ks < KS; ++ks) {
    const int buf = ks & 1;

    if (ks + 1 < KS) {
      // other buffer is reader-free (previous barrier); stream next tile in
      async_copy_32B(buf ? ldsA0 : ldsA1, gA + (size_t)(ks + 1) * BK);
      async_copy_32B(buf ? ldsB0 : ldsB1, gB + (size_t)(ks + 1) * BK);
    }

    Frag af[2], bfr[4];
#pragma unroll
    for (int tm = 0; tm < 2; ++tm) {
      const bf16_t* p = &sA[buf][(wm * 32 + tm * 16 + fr) * LDSK + kb];
      af[tm].h[0] = *reinterpret_cast<const v8bf*>(p);
      af[tm].h[1] = *reinterpret_cast<const v8bf*>(p + 16);
    }
#pragma unroll
    for (int tn = 0; tn < 4; ++tn) {
      const bf16_t* p = &sB[buf][(wn * 64 + tn * 16 + fr) * LDSK + kb];
      bfr[tn].h[0] = *reinterpret_cast<const v8bf*>(p);
      bfr[tn].h[1] = *reinterpret_cast<const v8bf*>(p + 16);
    }
#pragma unroll
    for (int tm = 0; tm < 2; ++tm)
#pragma unroll
      for (int tn = 0; tn < 4; ++tn)
        acc[tm][tn] = wmma_bf16(af[tm], bfr[tn], acc[tm][tn]);

    wait_async0();
    __syncthreads();
  }

  // Epilogue: C/D layout -> lane<16: rows i, lane>=16: rows 8+i, col = base+(lane&15)
  const int mo = (lane >> 4) * 8;
#pragma unroll
  for (int tm = 0; tm < 2; ++tm) {
#pragma unroll
    for (int tn = 0; tn < 4; ++tn) {
      const int col = blockN + wn * 64 + tn * 16 + fr;
      const float bv = bias ? bias[col] : 0.0f;
#pragma unroll
      for (int i = 0; i < 8; ++i) {
        const int row = blockM + wm * 32 + tm * 16 + mo + i;
        const size_t idx = (size_t)row * N + col;
        float c = acc[tm][tn][i] + bv;
        if constexpr (EPI == EPI_F32) {
          outF[idx] = c;
        } else if constexpr (EPI == EPI_BF16) {
          outB[idx] = (bf16_t)c;
        } else if constexpr (EPI == EPI_RESID) {
          outF[idx] = resid[idx] + gamma[0] * c;
        } else {  // EPI_GELU (exact)
          outB[idx] = (bf16_t)(0.5f * c * (1.0f + erff(c * 0.70710678118f)));
        }
      }
    }
  }
}

// ---------------------------------------------------------------------------
// Row LayerNorm (cols = 2048) f32 in -> bf16 out. One row per 256-thread block.
// ---------------------------------------------------------------------------
__global__ __launch_bounds__(256)
void layernorm_bf16(const float* __restrict__ x, const float* __restrict__ g,
                    const float* __restrict__ b, bf16_t* __restrict__ y) {
  const size_t row = blockIdx.x;
  const int tid = threadIdx.x;
  const float* xr = x + row * DD;
  float local[8];
  float s = 0.f, s2 = 0.f;
#pragma unroll
  for (int j = 0; j < 8; ++j) {
    float v = xr[tid + j * 256];
    local[j] = v; s += v; s2 += v * v;
  }
  s = waveRedSum(s); s2 = waveRedSum(s2);
  __shared__ float rs[8], rs2[8];
  if ((tid & 31) == 0) { rs[tid >> 5] = s; rs2[tid >> 5] = s2; }
  __syncthreads();
  float ts = 0.f, ts2 = 0.f;
#pragma unroll
  for (int w = 0; w < 8; ++w) { ts += rs[w]; ts2 += rs2[w]; }
  const float mu  = ts * (1.0f / DD);
  const float inv = rsqrtf(ts2 * (1.0f / DD) - mu * mu + 1e-5f);
#pragma unroll
  for (int j = 0; j < 8; ++j) {
    int cidx = tid + j * 256;
    y[row * DD + cidx] = (bf16_t)((local[j] - mu) * inv * g[cidx] + b[cidx]);
  }
}

// ---------------------------------------------------------------------------
// Attention: one block per (b,h). S = QK^T/16 + relpos bias; softmax; Z = P V.
// Q/K fragments straight from global (16B/lane); V staged transposed in LDS.
// ---------------------------------------------------------------------------
__global__ __launch_bounds__(256)
void attention_kernel(const bf16_t* __restrict__ qkv, const float* __restrict__ relb,
                      bf16_t* __restrict__ out /* (B,L,D) bf16 */) {
  const int bh = blockIdx.x;
  const int b = bh >> 3, h = bh & 7;
  const size_t base = (size_t)b * LL * (3 * DD) + (size_t)h * DHH;  // q offset
  const int tid = threadIdx.x, lane = tid & 31, wave = tid >> 5;

  __shared__ float  sS[64 * 68];
  __shared__ bf16_t sP[64 * 72];
  __shared__ bf16_t sVT[256 * 72];  // [d][l]

  // stage V transposed: 16B coalesced global loads, scatter into sVT[d][l]
#pragma unroll
  for (int j = 0; j < 8; ++j) {
    int idx = tid + j * 256;          // 0..2047
    int l  = idx >> 5;                // 0..63
    int d0 = (idx & 31) * 8;          // 0..248
    const v8bf vv = *reinterpret_cast<const v8bf*>(
        &qkv[base + (size_t)l * (3 * DD) + 2 * DD + d0]);
#pragma unroll
    for (int e = 0; e < 8; ++e) sVT[(d0 + e) * 72 + l] = vv[e];
  }

  const int fr = lane & 15;
  const int kb = (lane >> 4) * 8;
  const int mo = (lane >> 4) * 8;

  // ---- S = Q @ K^T : 4x4 tiles of 16x16; wave w owns (tm = w>>1, tn0=(w&1)*2..+1)
  const int tm = wave >> 1, tn0 = (wave & 1) * 2;
  v8f zero = {};
  v8f acc[2] = {zero, zero};
  for (int kk = 0; kk < DHH; kk += 32) {
    Frag a;
    const bf16_t* qa = qkv + base + (size_t)(tm * 16 + fr) * (3 * DD) + kk + kb;
    a.h[0] = *reinterpret_cast<const v8bf*>(qa);
    a.h[1] = *reinterpret_cast<const v8bf*>(qa + 16);
#pragma unroll
    for (int t = 0; t < 2; ++t) {
      Frag kf;
      const bf16_t* kp = qkv + base + (size_t)((tn0 + t) * 16 + fr) * (3 * DD) + DD + kk + kb;
      kf.h[0] = *reinterpret_cast<const v8bf*>(kp);
      kf.h[1] = *reinterpret_cast<const v8bf*>(kp + 16);
      acc[t] = wmma_bf16(a, kf, acc[t]);
    }
  }
  // scale + rel-pos bias -> LDS (f32)
#pragma unroll
  for (int t = 0; t < 2; ++t) {
    const int n = (tn0 + t) * 16 + fr;
#pragma unroll
    for (int i = 0; i < 8; ++i) {
      const int m = tm * 16 + mo + i;
      const int ri = ((m >> 3) - (n >> 3) + 7) * 15 + ((m & 7) - (n & 7) + 7);
      sS[m * 68 + n] = acc[t][i] * 0.0625f + relb[h * 225 + ri];
    }
  }
  __syncthreads();

  // ---- softmax: 4 threads per row, 16 cols each; shuffle within groups of 4
  {
    const int r = tid >> 2, c0 = (tid & 3) * 16;
    float mx = -1e30f;
#pragma unroll
    for (int j = 0; j < 16; ++j) mx = fmaxf(mx, sS[r * 68 + c0 + j]);
#pragma unroll
    for (int o = 1; o < 4; o <<= 1) mx = fmaxf(mx, __shfl_xor(mx, o, 32));
    float e[16], sm = 0.f;
#pragma unroll
    for (int j = 0; j < 16; ++j) { e[j] = __expf(sS[r * 68 + c0 + j] - mx); sm += e[j]; }
#pragma unroll
    for (int o = 1; o < 4; o <<= 1) sm += __shfl_xor(sm, o, 32);
    const float inv = 1.0f / sm;
#pragma unroll
    for (int j = 0; j < 16; ++j) sP[r * 72 + c0 + j] = (bf16_t)(e[j] * inv);
  }
  __syncthreads();

  // ---- Z = P @ V : M=64 (4 tiles) x N=256 (16 tiles); wave owns (w>>1, 8 tiles)
  const int tmz = wave >> 1, tnb = (wave & 1) * 8;
  v8f zacc[8];
#pragma unroll
  for (int t = 0; t < 8; ++t) zacc[t] = zero;
  for (int kk = 0; kk < 64; kk += 32) {
    Frag a;
    const bf16_t* pa = &sP[(tmz * 16 + fr) * 72 + kk + kb];
    a.h[0] = *reinterpret_cast<const v8bf*>(pa);
    a.h[1] = *reinterpret_cast<const v8bf*>(pa + 16);
#pragma unroll
    for (int t = 0; t < 8; ++t) {
      Frag vf;
      const bf16_t* vp = &sVT[((tnb + t) * 16 + fr) * 72 + kk + kb];
      vf.h[0] = *reinterpret_cast<const v8bf*>(vp);
      vf.h[1] = *reinterpret_cast<const v8bf*>(vp + 16);
      zacc[t] = wmma_bf16(a, vf, zacc[t]);
    }
  }
#pragma unroll
  for (int t = 0; t < 8; ++t) {
    const int n = (tnb + t) * 16 + fr;
#pragma unroll
    for (int i = 0; i < 8; ++i) {
      const int m = tmz * 16 + mo + i;
      out[((size_t)b * LL + m) * DD + h * DHH + n] = (bf16_t)zacc[t][i];
    }
  }
}

// ---------------------------------------------------------------------------
// Attention pooling: per b — LN(tokens) · pool_query -> softmax over L ->
// pooled = sum_l w[l] * tokens[l, :]. One block per batch, 8 waves x 8 rows.
// ---------------------------------------------------------------------------
__global__ __launch_bounds__(256)
void pool_kernel(const float* __restrict__ tokens, const float* __restrict__ png,
                 const float* __restrict__ pnb, const float* __restrict__ pq,
                 float* __restrict__ pooledF, bf16_t* __restrict__ pooledB) {
  const int b = blockIdx.x;
  const int tid = threadIdx.x, lane = tid & 31, wave = tid >> 5;
  const float* tb = tokens + (size_t)b * LL * DD;
  __shared__ float logits[64];

  for (int r8 = 0; r8 < 8; ++r8) {
    const int l = wave * 8 + r8;
    const float* row = tb + (size_t)l * DD;
    float s = 0.f, s2 = 0.f;
#pragma unroll 8
    for (int j = 0; j < 64; ++j) { float v = row[lane + j * 32]; s += v; s2 += v * v; }
    s = waveRedSum(s); s2 = waveRedSum(s2);
    const float mu  = s * (1.0f / DD);
    const float inv = rsqrtf(s2 * (1.0f / DD) - mu * mu + 1e-5f);
    float dot = 0.f;
#pragma unroll 8
    for (int j = 0; j < 64; ++j) {
      const int d = lane + j * 32;
      const float kv = (row[d] - mu) * inv * png[d] + pnb[d];
      dot += kv * pq[d];
    }
    dot = waveRedSum(dot);
    if (lane == 0) logits[l] = dot * 0.02209708691f;  // 1/sqrt(2048)
  }
  __syncthreads();
  if (tid == 0) {
    float mx = -1e30f;
    for (int i = 0; i < 64; ++i) mx = fmaxf(mx, logits[i]);
    float sm = 0.f;
    for (int i = 0; i < 64; ++i) { logits[i] = __expf(logits[i] - mx); sm += logits[i]; }
    const float inv = 1.0f / sm;
    for (int i = 0; i < 64; ++i) logits[i] *= inv;
  }
  __syncthreads();
  for (int d = tid; d < DD; d += 256) {
    float s = 0.f;
#pragma unroll 8
    for (int l = 0; l < 64; ++l) s += logits[l] * tb[(size_t)l * DD + d];
    pooledF[(size_t)b * DD + d] = s;
    pooledB[(size_t)b * DD + d] = (bf16_t)s;
  }
}

// ---------------------------------------------------------------------------
// Final per-row L2 normalize: out = p2 / max(||p2||, 1e-12). One block per b.
// ---------------------------------------------------------------------------
__global__ __launch_bounds__(256)
void norm_out_kernel(const float* __restrict__ p2, float* __restrict__ out) {
  const int b = blockIdx.x, tid = threadIdx.x;
  const float* r = p2 + (size_t)b * DD;
  float s = 0.f;
#pragma unroll
  for (int j = 0; j < 8; ++j) { float v = r[tid + j * 256]; s += v * v; }
  s = waveRedSum(s);
  __shared__ float red[8];
  if ((tid & 31) == 0) red[tid >> 5] = s;
  __syncthreads();
  float tot = 0.f;
#pragma unroll
  for (int w = 0; w < 8; ++w) tot += red[w];
  const float inv = 1.0f / fmaxf(sqrtf(tot), 1e-12f);
#pragma unroll
  for (int j = 0; j < 8; ++j) {
    const int i = tid + j * 256;
    out[(size_t)b * DD + i] = r[i] * inv;
  }
}

// ---------------------------------------------------------------------------
// Host-side launcher
// ---------------------------------------------------------------------------
extern "C" void kernel_launch(void* const* d_in, const int* in_sizes, int n_in,
                              void* d_out, int out_size, void* d_ws, size_t ws_size,
                              hipStream_t stream) {
  (void)in_sizes; (void)n_in; (void)out_size; (void)ws_size;
  const float* x      = (const float*)d_in[0];
  const float* in_w   = (const float*)d_in[1];
  const float* in_b   = (const float*)d_in[2];
  const float* qkv_w  = (const float*)d_in[3];
  const float* qkv_b  = (const float*)d_in[4];
  const float* out_w  = (const float*)d_in[5];
  const float* out_b  = (const float*)d_in[6];
  const float* relb   = (const float*)d_in[7];
  const float* n1g    = (const float*)d_in[8];
  const float* n1b    = (const float*)d_in[9];
  const float* n2g    = (const float*)d_in[10];
  const float* n2b    = (const float*)d_in[11];
  const float* fc1_w  = (const float*)d_in[12];
  const float* fc1_b  = (const float*)d_in[13];
  const float* fc2_w  = (const float*)d_in[14];
  const float* fc2_b  = (const float*)d_in[15];
  const float* g_attn = (const float*)d_in[16];
  const float* g_mlp  = (const float*)d_in[17];
  const float* pq     = (const float*)d_in[18];
  const float* png    = (const float*)d_in[19];
  const float* pnb    = (const float*)d_in[20];
  const float* pool_w = (const float*)d_in[21];
  const float* pool_b = (const float*)d_in[22];
  float* outp = (float*)d_out;

  char* ws = (char*)d_ws;
  size_t off = 0;
  auto alloc = [&](size_t bytes) -> char* {
    char* p = ws + off;
    off += (bytes + 255) & ~(size_t)255;
    return p;
  };

  bf16_t* wbIn   = (bf16_t*)alloc((size_t)DD * CC * 2);
  bf16_t* wbQkv  = (bf16_t*)alloc((size_t)3 * DD * DD * 2);
  bf16_t* wbOut  = (bf16_t*)alloc((size_t)CC * DD * 2);
  bf16_t* wbFc1  = (bf16_t*)alloc((size_t)HIDD * DD * 2);
  bf16_t* wbFc2  = (bf16_t*)alloc((size_t)DD * HIDD * 2);
  bf16_t* wbPool = (bf16_t*)alloc((size_t)CC * DD * 2);
  bf16_t* xt      = (bf16_t*)alloc((size_t)MTOK * CC * 2);
  float*  tokens1 = (float*) alloc((size_t)MTOK * DD * 4);
  bf16_t* yb      = (bf16_t*)alloc((size_t)MTOK * DD * 2);
  bf16_t* qkvb    = (bf16_t*)alloc((size_t)MTOK * 3 * DD * 2);
  bf16_t* attnout = (bf16_t*)alloc((size_t)MTOK * DD * 2);
  float*  tokens2 = (float*) alloc((size_t)MTOK * DD * 4);
  bf16_t* hb      = (bf16_t*)alloc((size_t)MTOK * HIDD * 2);
  float*  tokens3 = (float*) alloc((size_t)MTOK * DD * 4);
  float*  pooledF = (float*) alloc((size_t)BB * DD * 4);
  bf16_t* pooledB = (bf16_t*)alloc((size_t)BB * DD * 2);
  float*  pooled2 = (float*) alloc((size_t)BB * DD * 4);

  auto castGrid = [](size_t n) -> unsigned {
    size_t g = (n + 255) / 256;
    return (unsigned)(g > 4096 ? 4096 : g);
  };

  // 0) weight casts f32 -> bf16
  cast_f32_bf16<<<castGrid((size_t)DD * CC), 256, 0, stream>>>(in_w, wbIn, (size_t)DD * CC);
  cast_f32_bf16<<<castGrid((size_t)3 * DD * DD), 256, 0, stream>>>(qkv_w, wbQkv, (size_t)3 * DD * DD);
  cast_f32_bf16<<<castGrid((size_t)CC * DD), 256, 0, stream>>>(out_w, wbOut, (size_t)CC * DD);
  cast_f32_bf16<<<castGrid((size_t)HIDD * DD), 256, 0, stream>>>(fc1_w, wbFc1, (size_t)HIDD * DD);
  cast_f32_bf16<<<castGrid((size_t)DD * HIDD), 256, 0, stream>>>(fc2_w, wbFc2, (size_t)DD * HIDD);
  cast_f32_bf16<<<castGrid((size_t)CC * DD), 256, 0, stream>>>(pool_w, wbPool, (size_t)CC * DD);

  // 1) x (B,C,8,8) -> xt (B*L, C) bf16
  transpose_x<<<dim3(CC / 64, BB), 256, 0, stream>>>(x, xt);

  // 2) tokens = xt @ in_proj_w^T + b  (f32 out)
  gemm_bf16_wmma<EPI_F32><<<dim3(DD / BN, MTOK / BM), 256, 0, stream>>>(
      xt, wbIn, in_b, nullptr, nullptr, tokens1, nullptr, MTOK, DD, CC);

  // 3) y = LN1(tokens) -> bf16
  layernorm_bf16<<<MTOK, 256, 0, stream>>>(tokens1, n1g, n1b, yb);

  // 4) qkv = y @ qkv_w^T + b  (bf16 out)
  gemm_bf16_wmma<EPI_BF16><<<dim3(3 * DD / BN, MTOK / BM), 256, 0, stream>>>(
      yb, wbQkv, qkv_b, nullptr, nullptr, nullptr, qkvb, MTOK, 3 * DD, DD);

  // 5) attention per (b,h)
  attention_kernel<<<BB * NHH, 256, 0, stream>>>(qkvb, relb, attnout);

  // 6) tokens2 = tokens1 + gamma_attn * (attnout @ out_proj_w^T + b)
  gemm_bf16_wmma<EPI_RESID><<<dim3(DD / BN, MTOK / BM), 256, 0, stream>>>(
      attnout, wbOut, out_b, tokens1, g_attn, tokens2, nullptr, MTOK, DD, DD);

  // 7) y2 = LN2(tokens2) -> bf16 (reuse yb)
  layernorm_bf16<<<MTOK, 256, 0, stream>>>(tokens2, n2g, n2b, yb);

  // 8) h = gelu(y2 @ fc1_w^T + b)  (bf16 out)
  gemm_bf16_wmma<EPI_GELU><<<dim3(HIDD / BN, MTOK / BM), 256, 0, stream>>>(
      yb, wbFc1, fc1_b, nullptr, nullptr, nullptr, hb, MTOK, HIDD, DD);

  // 9) tokens3 = tokens2 + gamma_mlp * (h @ fc2_w^T + b)
  gemm_bf16_wmma<EPI_RESID><<<dim3(DD / BN, MTOK / BM), 256, 0, stream>>>(
      hb, wbFc2, fc2_b, tokens2, g_mlp, tokens3, nullptr, MTOK, DD, HIDD);

  // 10) attention pooling over L
  pool_kernel<<<BB, 256, 0, stream>>>(tokens3, png, pnb, pq, pooledF, pooledB);

  // 11) pooled2 = pooled @ pool_proj_w^T + b
  gemm_bf16_wmma<EPI_F32><<<dim3(DD / BN, BB / BM), 256, 0, stream>>>(
      pooledB, wbPool, pool_b, nullptr, nullptr, pooled2, nullptr, BB, DD, DD);

  // 12) L2 normalize rows -> d_out
  norm_out_kernel<<<BB, 256, 0, stream>>>(pooled2, outp);
}
